// RNN_2448131359131
// MI455X (gfx1250) — matrix operand order repaired and checked by reading, not compiled, not run on hardware
//
#include <hip/hip_runtime.h>
#include <hip/hip_bf16.h>
#include <stdint.h>

// ---------------------------------------------------------------------------
// modReLU RNN on MI455X (gfx1250), wave32 + WMMA bf16.
//   B=64, T=1024, D=256, H=1024, C=1000
// Phase 1: xproj = x @ w_ih^T           (WMMA GEMM, stored bf16)
// Phase 2: 1024-step recurrence          (single persistent kernel; W_hh strip
//          staged in LDS via async-copy; device-wide barrier per step)
// Phase 3: out = h_last @ w_fc^T + b_fc  (WMMA GEMM, padded N)
// ---------------------------------------------------------------------------

#define BSZ   64
#define TLEN  1024
#define DDIM  256
#define HDIM  1024
#define CDIM  1000
#define CPAD  1008          // 63 * 16
#define RNN_BLOCKS 64       // persistent kernel blocks (4 waves each)
#define LDSPITCH 1032       // 1024 + 8 halfword pad: conflict-free ds_load_b128

#if defined(__has_builtin)
#if __has_builtin(__builtin_amdgcn_global_load_async_to_lds_b128) && \
    __has_builtin(__builtin_amdgcn_s_wait_asynccnt)
#define USE_ASYNC_LDS 1
#endif
#endif

typedef __attribute__((ext_vector_type(16))) __bf16          v16bf;
typedef __attribute__((ext_vector_type(8)))  float           v8f;
typedef __attribute__((ext_vector_type(8)))  unsigned short  ushort8;

// 16-byte int vector matching the async-copy builtin's pointee type
// (diagnostic showed: "__attribute__((__vector_size__(4 * sizeof(int)))) int")
typedef int v4i_b128 __attribute__((vector_size(16)));
typedef __attribute__((address_space(1))) v4i_b128* gptr_b128;   // global (AS1)
typedef __attribute__((address_space(3))) v4i_b128* lptr_b128;   // LDS (AS3)

union V16U { ushort8 u[2]; v16bf v; };

__device__ __forceinline__ unsigned short f32_to_bf16(float f) {
    unsigned int u = __builtin_bit_cast(unsigned int, f);
    u += 0x7FFFu + ((u >> 16) & 1u);          // round-to-nearest-even
    return (unsigned short)(u >> 16);
}
__device__ __forceinline__ float bf16_to_f32(unsigned short s) {
    unsigned int u = ((unsigned int)s) << 16;
    return __builtin_bit_cast(float, u);
}

// A fragment: 16x32 bf16 tile from row-major [rows x lda] matrix.
// lanes 0-15: M=lane, K = k0+0..15
// lanes16-31: M=lane-16, K = k0+16..31
__device__ __forceinline__ v16bf load_a_bf16(const unsigned short* __restrict__ A,
                                             int lda, int row0, int k0, int lane) {
    int m  = row0 + (lane & 15);
    int kb = k0 + ((lane >> 4) << 4);
    const unsigned short* p = A + (size_t)m * lda + kb;
    V16U r;
    r.u[0] = *(const ushort8*)p;          // K = kb .. kb+7
    r.u[1] = *(const ushort8*)(p + 8);    // K = kb+8 .. kb+15
    return r.v;
}

// B fragment: 32x16 bf16, B(k,n) = W[n, k] with W row-major [.. x ldw]
// (z = h * W^T => row-major W is exactly column-major B).
// lanes 0-15: N=lane, K = k0+0..15 ; lanes 16-31: N=lane-16, K = k0+16..31
__device__ __forceinline__ v16bf load_b_bf16(const unsigned short* __restrict__ W,
                                             int ldw, int n0, int k0, int lane) {
    int n  = n0 + (lane & 15);
    int kb = k0 + ((lane >> 4) << 4);
    const unsigned short* p = W + (size_t)n * ldw + kb;
    V16U r;
    r.u[0] = *(const ushort8*)p;
    r.u[1] = *(const ushort8*)(p + 8);
    return r.v;
}

// Same B fragment but from the padded LDS strip (n0 already factored out).
__device__ __forceinline__ v16bf load_b_lds(const unsigned short* Wlds,
                                            int k0, int lane) {
    int n  = lane & 15;
    int kb = k0 + ((lane >> 4) << 4);
    const unsigned short* p = Wlds + n * LDSPITCH + kb;
    V16U r;
    r.u[0] = *(const ushort8*)p;
    r.u[1] = *(const ushort8*)(p + 8);
    return r.v;
}

__device__ __forceinline__ v8f wmma_bf16(v16bf a, v16bf b, v8f c) {
    return __builtin_amdgcn_wmma_f32_16x16x32_bf16(
        /*neg_a=*/false, a, /*neg_b=*/false, b,
        /*c_mod=*/(short)0, c, /*reuse_a=*/false, /*reuse_b=*/false);
}

// ---------------------------------------------------------------------------
// fp32 -> bf16 conversion kernels
// ---------------------------------------------------------------------------
__global__ void cvt_f32_bf16_kernel(const float* __restrict__ in,
                                    unsigned short* __restrict__ out, int n) {
    int i = blockIdx.x * blockDim.x + threadIdx.x;
    if (i < n) out[i] = f32_to_bf16(in[i]);
}

__global__ void cvt_wfc_padded_kernel(const float* __restrict__ wfc,
                                      unsigned short* __restrict__ out) {
    int i = blockIdx.x * blockDim.x + threadIdx.x;   // over CPAD*HDIM
    if (i < CPAD * HDIM) {
        int row = i >> 10, col = i & (HDIM - 1);
        out[i] = (row < CDIM) ? f32_to_bf16(wfc[row * HDIM + col]) : (unsigned short)0;
    }
}

// ---------------------------------------------------------------------------
// Phase 1: xproj[bt, h] = sum_d x[bt, d] * w_ih[h, d], stored bf16.
// grid = (4096 M-tiles, 16 N-groups), block = 128 (4 waves, 1 N-tile each)
// ---------------------------------------------------------------------------
__global__ void xproj_gemm_kernel(const unsigned short* __restrict__ xb,   // [B*T, D]
                                  const unsigned short* __restrict__ wih,  // [H, D]
                                  unsigned short* __restrict__ xproj) {    // [B*T, H]
    const int lane = threadIdx.x & 31;
    const int wave = threadIdx.x >> 5;
    const int row0 = blockIdx.x * 16;
    const int n0   = (blockIdx.y * 4 + wave) * 16;
    v8f c0 = {}, c1 = {};
#pragma unroll
    for (int kt = 0; kt < DDIM / 32; kt += 2) {
        v16bf a0 = load_a_bf16(xb,  DDIM, row0, kt * 32, lane);
        v16bf b0 = load_b_bf16(wih, DDIM, n0,   kt * 32, lane);
        c0 = wmma_bf16(a0, b0, c0);
        v16bf a1 = load_a_bf16(xb,  DDIM, row0, kt * 32 + 32, lane);
        v16bf b1 = load_b_bf16(wih, DDIM, n0,   kt * 32 + 32, lane);
        c1 = wmma_bf16(a1, b1, c1);
    }
    const int nc = n0 + (lane & 15);
#pragma unroll
    for (int r = 0; r < 8; ++r) {
        int m = row0 + r + ((lane >> 4) << 3);
        xproj[(size_t)m * HDIM + nc] = f32_to_bf16(c0[r] + c1[r]);
    }
}

// ---------------------------------------------------------------------------
// Phase 2: persistent recurrence. 64 blocks x 4 waves = 256 waves, each owns
// one 16x16 tile of the [64 x 1024] hidden state. All 4 waves of a block share
// the same 16-row W_hh strip (ntile = blockIdx.x), so it is staged once into
// LDS (async-copy when available) and re-read 1024x via ds_load_b128 with a
// bank-conflict-free padded pitch. Per step: 32 WMMA over K=1024 on two
// interleaved accumulators, add xproj_t, modReLU, store bf16 into the
// ping-pong buffer, then a device-wide barrier (fresh counter per step).
// ---------------------------------------------------------------------------
__global__ void rnn_recurrence_kernel(const unsigned short* __restrict__ xproj, // [B,T,H] bf16
                                      const unsigned short* __restrict__ whh,   // [H,H] bf16
                                      const float* __restrict__ b_mod,          // [H]
                                      unsigned short* __restrict__ hbuf0,       // [B,H] bf16 (zeroed)
                                      unsigned short* __restrict__ hbuf1,       // [B,H] bf16
                                      int* __restrict__ bar) {
    __shared__ unsigned short lds_w[16 * LDSPITCH];   // 33,024 B: one W_hh strip

    const int lane  = threadIdx.x & 31;
    const int wave  = threadIdx.x >> 5;
    const int row0  = wave * 16;               // batch tile (4 waves cover 64 rows)
    const int n0    = blockIdx.x * 16;         // hidden tile, shared by the block
    const int nc    = n0 + (lane & 15);
    const float bias = b_mod[nc];

    // ---- stage the block's 16 x 1024 W_hh strip into LDS (once) ----
    {
        const unsigned short* src = whh + (size_t)n0 * HDIM;
        const int tid = threadIdx.x;
#ifdef USE_ASYNC_LDS
#pragma unroll 4
        for (int row = 0; row < 16; ++row) {
            const unsigned short* s = src + row * HDIM + tid * 8;
            unsigned short*       d = &lds_w[row * LDSPITCH + tid * 8];
            __builtin_amdgcn_global_load_async_to_lds_b128(
                (gptr_b128)s, (lptr_b128)d, 0, 0);
        }
        __builtin_amdgcn_s_wait_asynccnt(0);
        __syncthreads();
#else
#pragma unroll 4
        for (int row = 0; row < 16; ++row) {
            *(ushort8*)&lds_w[row * LDSPITCH + tid * 8] =
                *(const ushort8*)(src + row * HDIM + tid * 8);
        }
        __syncthreads();
#endif
    }

    for (int t = 0; t < TLEN; ++t) {
        const unsigned short* hcur  = (t & 1) ? hbuf1 : hbuf0;
        unsigned short*       hnext = (t & 1) ? hbuf0 : hbuf1;

        v8f c0 = {}, c1 = {};
#pragma unroll 4
        for (int kt = 0; kt < HDIM / 32; kt += 2) {
            v16bf a0 = load_a_bf16(hcur, HDIM, row0, kt * 32, lane);
            v16bf b0 = load_b_lds(lds_w, kt * 32, lane);
            c0 = wmma_bf16(a0, b0, c0);
            v16bf a1 = load_a_bf16(hcur, HDIM, row0, kt * 32 + 32, lane);
            v16bf b1 = load_b_lds(lds_w, kt * 32 + 32, lane);
            c1 = wmma_bf16(a1, b1, c1);
        }

#pragma unroll
        for (int r = 0; r < 8; ++r) {
            int m   = row0 + r + ((lane >> 4) << 3);          // batch row
            float z = c0[r] + c1[r]
                    + bf16_to_f32(xproj[((size_t)m * TLEN + t) * HDIM + nc]);
            float mag = fabsf(z) + bias;
            float hv  = (mag > 0.0f) ? copysignf(mag, z) : 0.0f;
            hnext[(size_t)m * HDIM + nc] = f32_to_bf16(hv);
            // warm L2 for next step's xproj tile while we sit in the barrier
            if (t + 1 < TLEN)
                __builtin_prefetch(&xproj[((size_t)m * TLEN + t + 1) * HDIM + nc], 0, 1);
        }

        // device-wide barrier: one fresh counter per timestep
        __threadfence();
        __syncthreads();
        if (threadIdx.x == 0) {
            __hip_atomic_fetch_add(&bar[t], 1, __ATOMIC_ACQ_REL, __HIP_MEMORY_SCOPE_AGENT);
            while (__hip_atomic_load(&bar[t], __ATOMIC_ACQUIRE, __HIP_MEMORY_SCOPE_AGENT)
                   < RNN_BLOCKS) {
                __builtin_amdgcn_s_sleep(2);
            }
        }
        __syncthreads();
    }
}

// ---------------------------------------------------------------------------
// Phase 3: out[64,1000] = h_last @ w_fc^T + b_fc  (w_fc padded to 1008 rows)
// grid = 63 N-tiles, block = 128 (4 waves = 4 M-tiles covering all 64 rows)
// ---------------------------------------------------------------------------
__global__ void fc_head_kernel(const unsigned short* __restrict__ hlast, // [B,H] bf16
                               const unsigned short* __restrict__ wfc,   // [CPAD,H] bf16
                               const float* __restrict__ bfc,            // [C]
                               float* __restrict__ out) {                // [B,C]
    const int lane = threadIdx.x & 31;
    const int wave = threadIdx.x >> 5;
    const int row0 = wave * 16;
    const int n0   = blockIdx.x * 16;
    v8f c0 = {}, c1 = {};
#pragma unroll 4
    for (int kt = 0; kt < HDIM / 32; kt += 2) {
        v16bf a0 = load_a_bf16(hlast, HDIM, row0, kt * 32, lane);
        v16bf b0 = load_b_bf16(wfc,   HDIM, n0,   kt * 32, lane);
        c0 = wmma_bf16(a0, b0, c0);
        v16bf a1 = load_a_bf16(hlast, HDIM, row0, kt * 32 + 32, lane);
        v16bf b1 = load_b_bf16(wfc,   HDIM, n0,   kt * 32 + 32, lane);
        c1 = wmma_bf16(a1, b1, c1);
    }
    const int nc = n0 + (lane & 15);
    const float bias = (nc < CDIM) ? bfc[nc] : 0.0f;
#pragma unroll
    for (int r = 0; r < 8; ++r) {
        int m = row0 + r + ((lane >> 4) << 3);
        if (nc < CDIM) out[(size_t)m * CDIM + nc] = c0[r] + c1[r] + bias;
    }
}

// ---------------------------------------------------------------------------
// Host side: workspace layout (bytes, all 256-aligned)
//   [0)        bar:      1024 * 4                  =   4096
//   [4096)     hbuf0:    64*1024*2                 = 131072
//   [135168)   hbuf1:    64*1024*2                 = 131072   (memset: first 266240)
//   [266240)   whh_bf:   1024*1024*2               = 2097152
//   [2363392)  wih_bf:   1024*256*2                = 524288
//   [2887680)  wfc_bf:   1008*1024*2               = 2064384
//   [4952064)  x_bf:     64*1024*256*2             = 33554432
//   [38506496) xproj_bf: 64*1024*1024*2            = 134217728
//   total ~165 MB
// ---------------------------------------------------------------------------
extern "C" void kernel_launch(void* const* d_in, const int* in_sizes, int n_in,
                              void* d_out, int out_size, void* d_ws, size_t ws_size,
                              hipStream_t stream) {
    const float* x     = (const float*)d_in[0];   // [B,T,D]
    const float* w_ih  = (const float*)d_in[1];   // [H,D]
    const float* w_hh  = (const float*)d_in[2];   // [H,H]
    const float* b_mod = (const float*)d_in[3];   // [H]
    const float* w_fc  = (const float*)d_in[4];   // [C,H]
    const float* b_fc  = (const float*)d_in[5];   // [C]
    float* out = (float*)d_out;

    char* ws = (char*)d_ws;
    int*            bar     = (int*)(ws + 0);
    unsigned short* hbuf0   = (unsigned short*)(ws + 4096);
    unsigned short* hbuf1   = (unsigned short*)(ws + 135168);
    unsigned short* whh_bf  = (unsigned short*)(ws + 266240);
    unsigned short* wih_bf  = (unsigned short*)(ws + 2363392);
    unsigned short* wfc_bf  = (unsigned short*)(ws + 2887680);
    unsigned short* x_bf    = (unsigned short*)(ws + 4952064);
    unsigned short* xproj   = (unsigned short*)(ws + 38506496);

    // zero barrier counters + both h ping-pong buffers (h0 must start at 0)
    (void)hipMemsetAsync(d_ws, 0, 266240, stream);

    // fp32 -> bf16 conversions
    {
        int n = BSZ * TLEN * DDIM;   // x
        cvt_f32_bf16_kernel<<<(n + 255) / 256, 256, 0, stream>>>(x, x_bf, n);
    }
    {
        int n = HDIM * DDIM;         // w_ih
        cvt_f32_bf16_kernel<<<(n + 255) / 256, 256, 0, stream>>>(w_ih, wih_bf, n);
    }
    {
        int n = HDIM * HDIM;         // w_hh
        cvt_f32_bf16_kernel<<<(n + 255) / 256, 256, 0, stream>>>(w_hh, whh_bf, n);
    }
    {
        int n = CPAD * HDIM;         // w_fc (zero-padded rows 1000..1007)
        cvt_wfc_padded_kernel<<<(n + 255) / 256, 256, 0, stream>>>(w_fc, wfc_bf);
    }

    // Phase 1: input projection GEMM
    xproj_gemm_kernel<<<dim3((BSZ * TLEN) / 16, HDIM / 64), 128, 0, stream>>>(
        x_bf, wih_bf, xproj);

    // Phase 2: 1024-step persistent recurrence
    rnn_recurrence_kernel<<<RNN_BLOCKS, 128, 0, stream>>>(
        xproj, whh_bf, b_mod, hbuf0, hbuf1, bar);

    // Phase 3: classifier head; after 1024 steps h_last lives in hbuf0
    fc_head_kernel<<<CPAD / 16, 128, 0, stream>>>(hbuf0, wfc_bf, b_fc, out);

    (void)in_sizes; (void)n_in; (void)out_size; (void)ws_size;
}
